// MambaMinimal_2293512536645
// MI455X (gfx1250) — compile-verified, matrix operand order checked by reading, not run on hardware
//
#include <hip/hip_runtime.h>
#include <hip/hip_bf16.h>

typedef __bf16 bf16_t;
typedef __attribute__((ext_vector_type(16))) __bf16 v16bf;
typedef __attribute__((ext_vector_type(8)))  __bf16 v8bf;
typedef __attribute__((ext_vector_type(8)))  float  v8f;

// TDM descriptor vector types (6-arg builtin: v4u, v8i, v4i, v4i, v8i, i32)
typedef __attribute__((ext_vector_type(4))) unsigned int tdm_v4u;
typedef __attribute__((ext_vector_type(8))) int          tdm_v8i;
typedef __attribute__((ext_vector_type(4))) int          tdm_v4i;

#define D_MODEL 1024
#define D_STATE 16
#define D_CONV  4
#define D_INNER 2048
#define DT_RANK 64
#define BATCH   2
#define SEQ     2048
#define MTOT    (BATCH * SEQ)            /* 4096 */
#define NSSM    (DT_RANK + 2 * D_STATE)  /* 96 */

// LDS tile geometry: 128 rows x 32 K of bf16; 64B payload padded to 80B rows
// (20 DWORDs: 16-lane b128 reads hit distinct bank quads, 16B aligned).
// The TDM produces this layout in hardware: pad_interval=3 (16 DWORDs),
// pad_amount=3 (4 DWORDs) -> +16B after every 64B.
#define LROW_E   40
#define LROW_B   80
#define LDS_TILE (128 * LROW_E)   /* 5120 elements = 10240 bytes */

// ---------------------------------------------------------------------------
// CDNA5 async global->LDS copy (ASYNCcnt) via inline asm (used in ssm_proj).
// ---------------------------------------------------------------------------
__device__ __forceinline__ void async_b128(unsigned lds_off, const void* gaddr) {
    asm volatile("global_load_async_to_lds_b128 %0, %1, off"
                 :: "v"(lds_off), "v"(gaddr) : "memory");
}
__device__ __forceinline__ void wait_async0() {
    asm volatile("s_wait_asynccnt 0x0" ::: "memory");
}

// ---------------------------------------------------------------------------
// TDM: DMA one 128 x 32 bf16 tile (row stride K elements) into LDS with
// hardware row padding 64B -> 80B.  D# built per ISA 08 §8.3/§8.4.
// ---------------------------------------------------------------------------
__device__ __forceinline__ void tdm_load_tile(unsigned lds_addr, const bf16_t* gptr, int K) {
    unsigned long long ga = (unsigned long long)(uintptr_t)gptr;
    tdm_v4u g0;
    g0[0] = 1u;                                   // count=1 (valid), user mode
    g0[1] = lds_addr;                             // LDS byte address
    g0[2] = (unsigned)(ga & 0xFFFFFFFFu);         // global_addr[31:0]
    g0[3] = (unsigned)((ga >> 32) & 0x01FFFFFFu)  // global_addr[56:32]
          | (2u << 30);                           // type=2 ("image")
    tdm_v8i g1;
    g1[0] = (int)((1u << 16)      /* data_size = 2 bytes            */
                | (1u << 20)      /* pad_enable                     */
                | (3u << 22)      /* pad_interval: 16 DWORDs (64B)  */
                | (3u << 25));    /* pad_amount: 4 DWORDs (16B)     */
    g1[1] = (int)(((unsigned)K & 0xFFFFu) << 16); // tensor_dim0[15:0] (abar=0)
    g1[2] = (int)((((unsigned)K >> 16) & 0xFFFFu) // tensor_dim0[31:16]
                | (128u << 16));                  // tensor_dim1 = 128 rows
    g1[3] = (int)(32u << 16);                     // tile_dim0 = 32 elements
    g1[4] = (int)128u;                            // tile_dim1 = 128 (tile_dim2=0)
    g1[5] = (int)(unsigned)K;                     // tensor_dim0_stride[31:0]
    g1[6] = 0;                                    // stride hi / dim1_stride lo
    g1[7] = 0;                                    // dim1_stride hi
    tdm_v4i gz4 = {0, 0, 0, 0};                   // groups 2/3: 2-D tensor
    tdm_v8i gz8 = {0, 0, 0, 0, 0, 0, 0, 0};
    __builtin_amdgcn_tensor_load_to_lds(g0, g1, gz4, gz4, gz8, 0);
}

// ---------------------------------------------------------------------------
// fp32 -> bf16 elementwise convert
// ---------------------------------------------------------------------------
__global__ void __launch_bounds__(256) cvt_bf16_kernel(const float* __restrict__ in,
                                                       bf16_t* __restrict__ out, int n) {
    int i = blockIdx.x * 256 + threadIdx.x;
    if (i < n) out[i] = (bf16_t)in[i];
}

// ---------------------------------------------------------------------------
// Fragment load (lo 16B + hi 16B), CDNA5 16-bit A/B layout:
// lanes 0-15 hold K{0..7,16..23}, lanes 16-31 hold K{8..15,24..31}.
// ---------------------------------------------------------------------------
__device__ __forceinline__ v16bf load_frag(const bf16_t* p) {
    v8bf lo = *(const v8bf*)(p);
    v8bf hi = *(const v8bf*)(p + 16);
    return __builtin_shufflevector(lo, hi, 0,1,2,3,4,5,6,7,8,9,10,11,12,13,14,15);
}

// ---------------------------------------------------------------------------
// Block-tiled WMMA bf16 GEMM:  C[M,N] f32 = A[M,K] bf16 rm * B[N,K]^T bf16 rm.
// 256 threads = 8 waves in 4(M) x 2(N); block tile 128x128; wave tile 32x64.
// A/B 128x32 K-slabs double-buffered in LDS, staged by the TDM (wave 0 issues,
// TENSORcnt-tracked, one workgroup barrier per K-step publishes the buffer).
// ---------------------------------------------------------------------------
__global__ void __launch_bounds__(256) wmma_gemm_tn(const bf16_t* __restrict__ A,
                                                    const bf16_t* __restrict__ B,
                                                    float* __restrict__ C,
                                                    int M, int N, int K) {
    __shared__ bf16_t smem[2][2][LDS_TILE];   // [buf][A=0/B=1][tile]

    const int tid  = threadIdx.x;
    const int wave = tid >> 5;
    const int lane = tid & 31;
    const int hl   = lane >> 4;
    const int r16  = lane & 15;
    const int wm   = wave >> 1;      // 0..3  -> 32-row stripe
    const int wn   = wave & 1;       // 0..1  -> 64-col stripe

    const int bn  = blockIdx.x % (N >> 7);
    const int bm  = blockIdx.x / (N >> 7);
    const int m0  = bm << 7;
    const int n0  = bn << 7;

    const unsigned lbase = (unsigned)(uintptr_t)(&smem[0][0][0]);
    const bf16_t* gA = A + (size_t)m0 * K;
    const bf16_t* gB = B + (size_t)n0 * K;

    v8f acc[2][4];
#pragma unroll
    for (int i = 0; i < 2; ++i)
#pragma unroll
        for (int j = 0; j < 4; ++j) acc[i][j] = (v8f){};

    // prologue: DMA buffer 0
    if (wave == 0) {
        tdm_load_tile(lbase,                          gA, K);
        tdm_load_tile(lbase + LDS_TILE * 2u,          gB, K);
        __builtin_amdgcn_s_wait_tensorcnt(0);
    }
    __syncthreads();

    for (int k0 = 0; k0 < K; k0 += 32) {
        const int buf = (k0 >> 5) & 1;

        // DMA next slab into the other buffer (safe: everyone passed the
        // barrier that followed their last read of it)
        if (wave == 0 && k0 + 32 < K) {
            const unsigned lshift = (unsigned)((buf ^ 1) * 2 * LDS_TILE * sizeof(bf16_t));
            tdm_load_tile(lbase + lshift,                 gA + (k0 + 32), K);
            tdm_load_tile(lbase + lshift + LDS_TILE * 2u, gB + (k0 + 32), K);
        }

        // ---- compute from LDS: 2 A-frags, 4 B-frags, 8 WMMAs ----
        const bf16_t* la = &smem[buf][0][(wm * 32 + r16) * LROW_E + hl * 8];
        const bf16_t* lb = &smem[buf][1][(wn * 64 + r16) * LROW_E + hl * 8];
        v16bf a0  = load_frag(la);
        v16bf a1  = load_frag(la + 16 * LROW_E);
        v16bf bf0 = load_frag(lb);
        v16bf bf1 = load_frag(lb + 16 * LROW_E);
        v16bf bf2 = load_frag(lb + 32 * LROW_E);
        v16bf bf3 = load_frag(lb + 48 * LROW_E);
        acc[0][0] = __builtin_amdgcn_wmma_f32_16x16x32_bf16(false, a0, false, bf0, (short)0, acc[0][0], false, false);
        acc[1][0] = __builtin_amdgcn_wmma_f32_16x16x32_bf16(false, a1, false, bf0, (short)0, acc[1][0], false, true);
        acc[0][1] = __builtin_amdgcn_wmma_f32_16x16x32_bf16(false, a0, false, bf1, (short)0, acc[0][1], false, false);
        acc[1][1] = __builtin_amdgcn_wmma_f32_16x16x32_bf16(false, a1, false, bf1, (short)0, acc[1][1], false, true);
        acc[0][2] = __builtin_amdgcn_wmma_f32_16x16x32_bf16(false, a0, false, bf2, (short)0, acc[0][2], false, false);
        acc[1][2] = __builtin_amdgcn_wmma_f32_16x16x32_bf16(false, a1, false, bf2, (short)0, acc[1][2], false, true);
        acc[0][3] = __builtin_amdgcn_wmma_f32_16x16x32_bf16(false, a0, false, bf3, (short)0, acc[0][3], false, false);
        acc[1][3] = __builtin_amdgcn_wmma_f32_16x16x32_bf16(false, a1, false, bf3, (short)0, acc[1][3], false, true);

        // publish next buffer
        if (wave == 0) __builtin_amdgcn_s_wait_tensorcnt(0);
        __syncthreads();
    }

    // ---- store: C/D layout VGPR i -> M = i + hl*8 within each 16-tile ----
#pragma unroll
    for (int it = 0; it < 2; ++it) {
#pragma unroll
        for (int j = 0; j < 4; ++j) {
            float* crow = C + (size_t)(m0 + wm * 32 + it * 16 + hl * 8) * N
                            + n0 + wn * 64 + j * 16 + r16;
#pragma unroll
            for (int i = 0; i < 8; ++i) crow[(size_t)i * N] = acc[it][j][i];
        }
    }
}

// ---------------------------------------------------------------------------
// Depthwise causal conv1d (width 4) + bias + SiLU.
// ---------------------------------------------------------------------------
__global__ void __launch_bounds__(256) conv_silu_kernel(const float* __restrict__ xz,
                                                        const float* __restrict__ conv_w,
                                                        const float* __restrict__ conv_b,
                                                        float* __restrict__ xc) {
    int i = blockIdx.x * 256 + threadIdx.x;
    if (i >= MTOT * D_INNER) return;
    int d = i & (D_INNER - 1);
    int m = i >> 11;
    int l = m & (SEQ - 1);
    int b = m >> 11;
    float acc = conv_b[d];
#pragma unroll
    for (int j = 0; j < D_CONV; ++j) {
        int lt = l - (D_CONV - 1) + j;
        if (lt >= 0)
            acc += xz[(size_t)(b * SEQ + lt) * (2 * D_INNER) + d] * conv_w[d * D_CONV + j];
    }
    xc[i] = acc / (1.0f + __expf(-acc));
}

// ---------------------------------------------------------------------------
// ssm = xc @ W_x^T : one block per token row; the 8KB row is staged into LDS
// with CDNA5 async global->LDS copies (ASYNCcnt).
// ---------------------------------------------------------------------------
__global__ void __launch_bounds__(128) ssm_proj_kernel(const float* __restrict__ xc,
                                                       const float* __restrict__ W_x,
                                                       float* __restrict__ ssm) {
    __shared__ float row[D_INNER];
    const int m = blockIdx.x;
    const unsigned lbase = (unsigned)(uintptr_t)(&row[0]);
    const float* src = xc + (size_t)m * D_INNER;
#pragma unroll
    for (int c = 0; c < 4; ++c) {
        const int e = (threadIdx.x + c * 128) * 4;        // 4 floats per chunk
        async_b128(lbase + (unsigned)(e * 4), src + e);
    }
    wait_async0();
    __syncthreads();
    const int r = threadIdx.x;
    if (r < NSSM) {
        const float* w = W_x + (size_t)r * D_INNER;
        float acc = 0.f;
        for (int k = 0; k < D_INNER; ++k) acc += row[k] * w[k];
        ssm[(size_t)m * NSSM + r] = acc;
    }
}

// ---------------------------------------------------------------------------
// dt = softplus(dt_low @ W_dt^T + b_dt)
// ---------------------------------------------------------------------------
__global__ void __launch_bounds__(256) dt_proj_kernel(const float* __restrict__ ssm,
                                                      const float* __restrict__ W_dt,
                                                      const float* __restrict__ b_dt,
                                                      float* __restrict__ dtb) {
    __shared__ float srow[DT_RANK];
    const int m = blockIdx.x;
    if (threadIdx.x < DT_RANK) srow[threadIdx.x] = ssm[(size_t)m * NSSM + threadIdx.x];
    __syncthreads();
    const int d = blockIdx.y * 256 + threadIdx.x;
    const float* w = W_dt + (size_t)d * DT_RANK;
    float acc = b_dt[d];
#pragma unroll 8
    for (int r = 0; r < DT_RANK; ++r) acc += srow[r] * w[r];
    dtb[(size_t)m * D_INNER + d] = (acc > 20.f) ? acc : log1pf(__expf(acc));
}

// ---------------------------------------------------------------------------
// Sequential selective scan, one thread per (b,d) channel; fuses skip + gate.
// ---------------------------------------------------------------------------
__global__ void __launch_bounds__(256) scan_gate_kernel(const float* __restrict__ dtb,
                                                        const float* __restrict__ ssm,
                                                        const float* __restrict__ xc,
                                                        const float* __restrict__ xz,
                                                        const float* __restrict__ A_log,
                                                        const float* __restrict__ Dp,
                                                        bf16_t* __restrict__ ygate) {
    const int idx = blockIdx.x * 256 + threadIdx.x;
    const int b = idx >> 11;
    const int d = idx & (D_INNER - 1);

    float Ad[D_STATE], h[D_STATE];
#pragma unroll
    for (int s = 0; s < D_STATE; ++s) {
        Ad[s] = -__expf(A_log[d * D_STATE + s]);
        h[s]  = 0.f;
    }
    const float Dd = Dp[d];

    for (int t = 0; t < SEQ; ++t) {
        const int m = b * SEQ + t;
        const float dtv = dtb[(size_t)m * D_INNER + d];
        const float xv  = xc [(size_t)m * D_INNER + d];
        const float zv  = xz [(size_t)m * (2 * D_INNER) + D_INNER + d];
        const float* Br = ssm + (size_t)m * NSSM + DT_RANK;
        const float* Cr = Br + D_STATE;
        float y = 0.f;
#pragma unroll
        for (int s = 0; s < D_STATE; ++s) {
            float dA = __expf(dtv * Ad[s]);
            h[s] = dA * h[s] + dtv * Br[s] * xv;
            y += h[s] * Cr[s];
        }
        y += Dd * xv;
        const float g = zv / (1.f + __expf(-zv));
        ygate[(size_t)m * D_INNER + d] = (bf16_t)(y * g);
    }
}

// ---------------------------------------------------------------------------
extern "C" void kernel_launch(void* const* d_in, const int* in_sizes, int n_in,
                              void* d_out, int out_size, void* d_ws, size_t ws_size,
                              hipStream_t stream) {
    const float* x      = (const float*)d_in[0];
    const float* W_in   = (const float*)d_in[1];
    const float* conv_w = (const float*)d_in[2];
    const float* conv_b = (const float*)d_in[3];
    const float* W_x    = (const float*)d_in[4];
    const float* W_dt   = (const float*)d_in[5];
    const float* b_dt   = (const float*)d_in[6];
    const float* A_log  = (const float*)d_in[7];
    const float* Dp     = (const float*)d_in[8];
    const float* W_out  = (const float*)d_in[9];
    float* out = (float*)d_out;

    char* ws = (char*)d_ws;
    size_t off = 0;
    auto carve = [&](size_t bytes) -> void* {
        void* p = (void*)(ws + off);
        off += (bytes + 255) & ~(size_t)255;
        return p;
    };
    bf16_t* xb    = (bf16_t*)carve((size_t)MTOT * D_MODEL * sizeof(bf16_t));
    bf16_t* Wib   = (bf16_t*)carve((size_t)2 * D_INNER * D_MODEL * sizeof(bf16_t));
    bf16_t* Wob   = (bf16_t*)carve((size_t)D_MODEL * D_INNER * sizeof(bf16_t));
    float*  xz    = (float*) carve((size_t)MTOT * 2 * D_INNER * sizeof(float));
    float*  xc    = (float*) carve((size_t)MTOT * D_INNER * sizeof(float));
    float*  ssm   = (float*) carve((size_t)MTOT * NSSM * sizeof(float));
    float*  dtb   = (float*) carve((size_t)MTOT * D_INNER * sizeof(float));
    bf16_t* ygate = (bf16_t*)carve((size_t)MTOT * D_INNER * sizeof(bf16_t));

    {
        int n = MTOT * D_MODEL;
        cvt_bf16_kernel<<<(n + 255) / 256, 256, 0, stream>>>(x, xb, n);
        n = 2 * D_INNER * D_MODEL;
        cvt_bf16_kernel<<<(n + 255) / 256, 256, 0, stream>>>(W_in, Wib, n);
        n = D_MODEL * D_INNER;
        cvt_bf16_kernel<<<(n + 255) / 256, 256, 0, stream>>>(W_out, Wob, n);
    }

    // 1) xz = x @ W_in^T   (4096 x 4096, K=1024): 32x32 blocks of 128x128
    wmma_gemm_tn<<<(MTOT / 128) * ((2 * D_INNER) / 128), 256, 0, stream>>>(
        xb, Wib, xz, MTOT, 2 * D_INNER, D_MODEL);

    // 2) depthwise conv + SiLU
    {
        int n = MTOT * D_INNER;
        conv_silu_kernel<<<(n + 255) / 256, 256, 0, stream>>>(xz, conv_w, conv_b, xc);
    }

    // 3) ssm projection, 4) dt projection + softplus
    ssm_proj_kernel<<<MTOT, 128, 0, stream>>>(xc, W_x, ssm);
    dt_proj_kernel<<<dim3(MTOT, D_INNER / 256), 256, 0, stream>>>(ssm, W_dt, b_dt, dtb);

    // 5) selective scan + skip + gate (emits bf16)
    scan_gate_kernel<<<(BATCH * D_INNER) / 256, 256, 0, stream>>>(
        dtb, ssm, xc, xz, A_log, Dp, ygate);

    // 6) out = ygate @ W_out^T  (4096 x 1024, K=2048): 32x8 blocks
    wmma_gemm_tn<<<(MTOT / 128) * (D_MODEL / 128), 256, 0, stream>>>(
        ygate, Wob, out, MTOT, D_MODEL, D_INNER);

    (void)in_sizes; (void)n_in; (void)out_size; (void)ws_size;
}
